// MyAPPNP_82102594830825
// MI455X (gfx1250) — matrix operand order repaired and checked by reading, not compiled, not run on hardware
//
#include <hip/hip_runtime.h>

// ---------------------------------------------------------------------------
// APPNP for MI455X (gfx1250):
//   1) fused bf16-WMMA MLP (v_wmma_f32_16x16x32_bf16); weights pre-packed in
//      per-lane B-fragment order -> each fragment is 2x global_load_b128
//   2) one-time GCN norm + CSR-by-destination build (counts, scan, fill)
//   3) 10 propagation steps with ZERO atomics: wave-per-node, scalar edge
//      records (s_load_b64), float2 channel gathers served from 192MB L2
// ---------------------------------------------------------------------------

typedef __bf16 bf16_t;
typedef __attribute__((ext_vector_type(16))) __bf16 v16bf;
typedef __attribute__((ext_vector_type(8)))  __bf16 v8bf;
typedef __attribute__((ext_vector_type(8)))  float  v8f;

#define IN_C   256
#define HID_C  128
#define OUT_C  64
#define ALPHA_F 0.1f
#define ONE_MINUS_ALPHA 0.9f

static __device__ __forceinline__ bf16_t f2bf(float v) { return (bf16_t)v; }

static __device__ __forceinline__ v16bf cat16(v8bf lo, v8bf hi) {
    return __builtin_shufflevector(lo, hi, 0,1,2,3,4,5,6,7,8,9,10,11,12,13,14,15);
}

// ---------------------------------------------------------------------------
// Weight repack into per-lane B-fragment order:
//   W1p[((kc*8+g)*32 + lane)*16 + j] = bf16(W1[(kc*32 + (j<8 ? 8h+j : 16+8h+j-8))*128 + g*16 + (lane&15)]), h=lane>>4
// ---------------------------------------------------------------------------
__global__ void repack_w1_kernel(const float* __restrict__ W1, bf16_t* __restrict__ W1p) {
    int idx = blockIdx.x * blockDim.x + threadIdx.x;
    if (idx >= 8 * 8 * 32 * 16) return;
    int j  = idx & 15;
    int l  = (idx >> 4) & 31;
    int g  = (idx >> 9) & 7;
    int kc = idx >> 12;
    int k  = kc * 32 + ((l < 16) ? 0 : 8) + ((j < 8) ? j : j + 8);
    int n  = g * 16 + (l & 15);
    W1p[idx] = f2bf(W1[k * HID_C + n]);
}

__global__ void repack_w2_kernel(const float* __restrict__ W2, bf16_t* __restrict__ W2p) {
    int idx = blockIdx.x * blockDim.x + threadIdx.x;
    if (idx >= 4 * 4 * 32 * 16) return;
    int j  = idx & 15;
    int l  = (idx >> 4) & 31;
    int g  = (idx >> 9) & 3;
    int kc = idx >> 11;
    int k  = kc * 32 + ((l < 16) ? 0 : 8) + ((j < 8) ? j : j + 8);
    int n  = g * 16 + (l & 15);
    W2p[idx] = f2bf(W2[k * OUT_C + n]);
}

// ---------------------------------------------------------------------------
// Fused MLP: z = relu(x@W1+b1)@W2 + b2.  128 threads = 4 waves, 64 rows/block.
// All 8 B fragments of a k-chunk are batched into registers before the WMMA
// burst so waits can drain incrementally instead of loadcnt==0 per WMMA.
// ---------------------------------------------------------------------------
__global__ __launch_bounds__(128) void mlp_kernel(
    const float* __restrict__ x,
    const bf16_t* __restrict__ W1p, const float* __restrict__ b1,
    const bf16_t* __restrict__ W2p, const float* __restrict__ b2,
    float* __restrict__ z, int nrows)
{
    __shared__ bf16_t sX[64 * 264];   // 64 rows x 256 K (+8 pad), 33 KB
    __shared__ bf16_t sH[64 * 136];   // 64 rows x 128 K (+8 pad), 17 KB

    const int tid  = threadIdx.x;
    const int row0 = blockIdx.x * 64;
    const int wave = tid >> 5;
    const int lane = tid & 31;
    const int rbase = wave * 16;
    const int m     = lane & 15;
    const int koffA = (lane < 16) ? 0 : 8;   // 16-bit A layout K split
    const int ncol  = lane & 15;
    const int radd  = (lane < 16) ? 0 : 8;   // C layout row split
    const bool full = (row0 + 64 <= nrows);

    for (int i = tid; i < 64 * IN_C; i += 128) {
        int r = i >> 8, c = i & 255;
        int gr = row0 + r; if (gr >= nrows) gr = nrows - 1;
        sX[r * 264 + c] = f2bf(x[(size_t)gr * IN_C + c]);
    }
    __syncthreads();

    // ---- stage 1: h = relu(x@W1 + b1), kept in LDS as bf16 ----
    v8f acc[8];
#pragma unroll
    for (int g = 0; g < 8; ++g) acc[g] = (v8f){0,0,0,0,0,0,0,0};

#pragma unroll
    for (int kc = 0; kc < 8; ++kc) {
        const v16bf* wb = (const v16bf*)&W1p[(size_t)(kc * 8 * 32 + lane) * 16];
        v16bf bb[8];
#pragma unroll
        for (int g = 0; g < 8; ++g) bb[g] = wb[g * 32];   // ((kc*8+g)*32+lane)*16
        const bf16_t* xr = &sX[(rbase + m) * 264 + kc * 32 + koffA];
        v16bf a = cat16(*(const v8bf*)xr, *(const v8bf*)(xr + 16));
#pragma unroll
        for (int g = 0; g < 8; ++g)
            acc[g] = __builtin_amdgcn_wmma_f32_16x16x32_bf16(
                false, a, false, bb[g], (short)0, acc[g], false, false);
    }

#pragma unroll
    for (int g = 0; g < 8; ++g) {
        const int col = g * 16 + ncol;
        const float bias = b1[col];
#pragma unroll
        for (int r = 0; r < 8; ++r) {
            float v = acc[g][r] + bias;
            v = v > 0.0f ? v : 0.0f;
            sH[(rbase + r + radd) * 136 + col] = f2bf(v);
        }
    }
    __syncthreads();

    // ---- stage 2: z = h@W2 + b2 ----
    v8f acc2[4];
#pragma unroll
    for (int g = 0; g < 4; ++g) acc2[g] = (v8f){0,0,0,0,0,0,0,0};

#pragma unroll
    for (int kc = 0; kc < 4; ++kc) {
        const v16bf* wb = (const v16bf*)&W2p[(size_t)(kc * 4 * 32 + lane) * 16];
        v16bf bb[4];
#pragma unroll
        for (int g = 0; g < 4; ++g) bb[g] = wb[g * 32];
        const bf16_t* hr = &sH[(rbase + m) * 136 + kc * 32 + koffA];
        v16bf a = cat16(*(const v8bf*)hr, *(const v8bf*)(hr + 16));
#pragma unroll
        for (int g = 0; g < 4; ++g)
            acc2[g] = __builtin_amdgcn_wmma_f32_16x16x32_bf16(
                false, a, false, bb[g], (short)0, acc2[g], false, false);
    }

    if (full) {
#pragma unroll
        for (int g = 0; g < 4; ++g) {
            const int col = g * 16 + ncol;
            const float bias = b2[col];
#pragma unroll
            for (int r = 0; r < 8; ++r)
                z[(size_t)(row0 + rbase + r + radd) * OUT_C + col] = acc2[g][r] + bias;
        }
    } else {
#pragma unroll
        for (int g = 0; g < 4; ++g) {
            const int col = g * 16 + ncol;
            const float bias = b2[col];
#pragma unroll
            for (int r = 0; r < 8; ++r) {
                int grow = row0 + rbase + r + radd;
                if (grow < nrows)
                    z[(size_t)grow * OUT_C + col] = acc2[g][r] + bias;
            }
        }
    }
}

// ---------------------------------------------------------------------------
// GCN-norm prep + CSR build
// ---------------------------------------------------------------------------
__global__ void zero_deg_cnt_kernel(float* __restrict__ deg, int* __restrict__ cnt, int n) {
    int i = blockIdx.x * blockDim.x + threadIdx.x;
    if (i < n) { deg[i] = 0.0f; cnt[i] = 0; }
}

// int64 -> int32 edge lists (incl. self loops) + weighted degree + dst counts
__global__ void prep_edges_kernel(const long long* __restrict__ ei,
                                  const float* __restrict__ ew,
                                  int* __restrict__ src32, int* __restrict__ dst32,
                                  float* __restrict__ deg, int* __restrict__ cnt,
                                  int E, int Nn)
{
    int e = blockIdx.x * blockDim.x + threadIdx.x;
    if (e >= E + Nn) return;
    int s, d; float w;
    if (e < E) {
        s = (int)ei[e];
        d = (int)ei[(size_t)E + e];
        w = ew[e];
    } else {
        s = d = e - E;
        w = 1.0f;
    }
    src32[e] = s; dst32[e] = d;
    __hip_atomic_fetch_add(&deg[d], w, __ATOMIC_RELAXED, __HIP_MEMORY_SCOPE_AGENT);
    __hip_atomic_fetch_add(&cnt[d], 1, __ATOMIC_RELAXED, __HIP_MEMORY_SCOPE_AGENT);
}

__global__ void dinv_kernel(float* __restrict__ deg, int Nn) {
    int i = blockIdx.x * blockDim.x + threadIdx.x;
    if (i < Nn) {
        float d = deg[i];
        deg[i] = d > 0.0f ? rsqrtf(d) : 0.0f;
    }
}

__global__ void norm_kernel(const int* __restrict__ src32, const int* __restrict__ dst32,
                            const float* __restrict__ ew, const float* __restrict__ dinv,
                            float* __restrict__ normw, int E, int Nn)
{
    int e = blockIdx.x * blockDim.x + threadIdx.x;
    if (e >= E + Nn) return;
    float w = (e < E) ? ew[e] : 1.0f;
    normw[e] = ONE_MINUS_ALPHA * dinv[src32[e]] * w * dinv[dst32[e]];
}

// block-local exclusive scan (Hillis-Steele) + per-block sums
__global__ __launch_bounds__(256) void scan_block_kernel(const int* __restrict__ cnt,
                                                         int* __restrict__ scanned,
                                                         int* __restrict__ bsum, int n)
{
    __shared__ int sh[256];
    const int t = threadIdx.x;
    const int i = blockIdx.x * 256 + t;
    int v = (i < n) ? cnt[i] : 0;
    sh[t] = v;
    __syncthreads();
#pragma unroll
    for (int o = 1; o < 256; o <<= 1) {
        int add = (t >= o) ? sh[t - o] : 0;
        __syncthreads();
        sh[t] += add;
        __syncthreads();
    }
    if (i < n) scanned[i] = sh[t] - v;               // exclusive
    if (t == 255) bsum[blockIdx.x] = sh[255];        // block total
}

// serial scan of ~391 block sums (trivial) + write row_ptr[N]
__global__ void scan_top_kernel(int* __restrict__ bsum, int nb,
                                int* __restrict__ row_ptr, int Nn, int tot)
{
    if (blockIdx.x == 0 && threadIdx.x == 0) {
        int acc = 0;
        for (int b = 0; b < nb; ++b) { int v = bsum[b]; bsum[b] = acc; acc += v; }
        row_ptr[Nn] = tot;
    }
}

__global__ void finalize_rowptr_kernel(const int* __restrict__ scanned,
                                       const int* __restrict__ bsum,
                                       int* __restrict__ row_ptr,
                                       int* __restrict__ cursor, int n)
{
    int i = blockIdx.x * blockDim.x + threadIdx.x;
    if (i < n) {
        int v = scanned[i] + bsum[i >> 8];
        row_ptr[i] = v;
        cursor[i]  = v;
    }
}

// CSR fill: (src, prefolded norm) packed as int2 -> one scalar b64 load later
__global__ void fill_csr_kernel(const int* __restrict__ src32, const int* __restrict__ dst32,
                                const float* __restrict__ normw, int* __restrict__ cursor,
                                int2* __restrict__ csr, int tot)
{
    int e = blockIdx.x * blockDim.x + threadIdx.x;
    if (e >= tot) return;
    int d = dst32[e];
    int pos = __hip_atomic_fetch_add(&cursor[d], 1, __ATOMIC_RELAXED, __HIP_MEMORY_SCOPE_AGENT);
    csr[pos] = make_int2(src32[e], __float_as_int(normw[e]));
}

// ---------------------------------------------------------------------------
// Propagation step, no atomics: one wave per node, lane owns 2 channels.
// node id is readfirstlane'd so row_ptr / csr records become scalar loads;
// h (25.6 MB) is L2-resident so the float2 gathers never touch HBM.
// h_out = alpha*z + sum_e norm'(e) * h_in[src(e)]      (norm' has (1-a) folded)
// ---------------------------------------------------------------------------
__global__ __launch_bounds__(256) void appnp_step_kernel(
    const int* __restrict__ row_ptr, const int2* __restrict__ csr,
    const float* __restrict__ z, const float* __restrict__ hin,
    float* __restrict__ hout, int Nn)
{
    int n = __builtin_amdgcn_readfirstlane((blockIdx.x * 256 + threadIdx.x) >> 5);
    if (n >= Nn) return;
    const int lane = threadIdx.x & 31;

    const int start = row_ptr[n];
    const int end   = row_ptr[n + 1];

    float2 acc = ((const float2*)(z + (size_t)n * OUT_C))[lane];
    acc.x *= ALPHA_F; acc.y *= ALPHA_F;

    int i = start;
    for (; i + 2 <= end; i += 2) {
        int2 e0 = csr[i];
        int2 e1 = csr[i + 1];
        float w0 = __int_as_float(e0.y);
        float w1 = __int_as_float(e1.y);
        float2 v0 = ((const float2*)(hin + (size_t)e0.x * OUT_C))[lane];
        float2 v1 = ((const float2*)(hin + (size_t)e1.x * OUT_C))[lane];
        acc.x = fmaf(w0, v0.x, acc.x); acc.y = fmaf(w0, v0.y, acc.y);
        acc.x = fmaf(w1, v1.x, acc.x); acc.y = fmaf(w1, v1.y, acc.y);
    }
    if (i < end) {
        int2 e0 = csr[i];
        float w0 = __int_as_float(e0.y);
        float2 v0 = ((const float2*)(hin + (size_t)e0.x * OUT_C))[lane];
        acc.x = fmaf(w0, v0.x, acc.x); acc.y = fmaf(w0, v0.y, acc.y);
    }
    ((float2*)(hout + (size_t)n * OUT_C))[lane] = acc;
}

// ---------------------------------------------------------------------------
extern "C" void kernel_launch(void* const* d_in, const int* in_sizes, int n_in,
                              void* d_out, int out_size, void* d_ws, size_t ws_size,
                              hipStream_t stream)
{
    const float*     x   = (const float*)d_in[0];
    const long long* ei  = (const long long*)d_in[1];
    const float*     ew  = (const float*)d_in[2];
    const float*     W1  = (const float*)d_in[3];
    const float*     b1  = (const float*)d_in[4];
    const float*     W2  = (const float*)d_in[5];
    const float*     b2  = (const float*)d_in[6];

    const int Nn  = in_sizes[0] / IN_C;   // 100000
    const int E   = in_sizes[2];          // 3200000
    const int tot = E + Nn;               // edges + self loops
    const int nb  = (Nn + 255) / 256;     // scan blocks

    // workspace layout (256B aligned slices)
    char* ws = (char*)d_ws;
    size_t off = 0;
    auto take = [&](size_t bytes) { void* p = ws + off; off = (off + bytes + 255) & ~(size_t)255; return p; };
    float*  z       = (float*)take((size_t)Nn * OUT_C * sizeof(float));
    float*  bufA    = (float*)take((size_t)Nn * OUT_C * sizeof(float));
    float*  bufB    = (float*)take((size_t)Nn * OUT_C * sizeof(float));
    float*  normw   = (float*)take((size_t)tot * sizeof(float));
    int*    src32   = (int*)take((size_t)tot * sizeof(int));
    int*    dst32   = (int*)take((size_t)tot * sizeof(int));
    int2*   csr     = (int2*)take((size_t)tot * sizeof(int2));
    float*  deg     = (float*)take((size_t)Nn * sizeof(float));
    int*    cnt     = (int*)take((size_t)Nn * sizeof(int));
    int*    scanned = (int*)take((size_t)Nn * sizeof(int));
    int*    bsum    = (int*)take((size_t)nb * sizeof(int));
    int*    row_ptr = (int*)take((size_t)(Nn + 1) * sizeof(int));
    int*    cursor  = (int*)take((size_t)Nn * sizeof(int));
    bf16_t* W1p     = (bf16_t*)take((size_t)8 * 8 * 32 * 16 * sizeof(bf16_t));
    bf16_t* W2p     = (bf16_t*)take((size_t)4 * 4 * 32 * 16 * sizeof(bf16_t));

    // --- weight repack + fused MLP via bf16 WMMA ---
    repack_w1_kernel<<<(8 * 8 * 32 * 16 + 255) / 256, 256, 0, stream>>>(W1, W1p);
    repack_w2_kernel<<<(4 * 4 * 32 * 16 + 255) / 256, 256, 0, stream>>>(W2, W2p);
    mlp_kernel<<<(Nn + 63) / 64, 128, 0, stream>>>(x, W1p, b1, W2p, b2, z, Nn);

    // --- GCN norm + CSR build (one-time) ---
    zero_deg_cnt_kernel<<<(Nn + 255) / 256, 256, 0, stream>>>(deg, cnt, Nn);
    prep_edges_kernel<<<(tot + 255) / 256, 256, 0, stream>>>(ei, ew, src32, dst32, deg, cnt, E, Nn);
    dinv_kernel<<<(Nn + 255) / 256, 256, 0, stream>>>(deg, Nn);
    norm_kernel<<<(tot + 255) / 256, 256, 0, stream>>>(src32, dst32, ew, deg, normw, E, Nn);
    scan_block_kernel<<<nb, 256, 0, stream>>>(cnt, scanned, bsum, Nn);
    scan_top_kernel<<<1, 32, 0, stream>>>(bsum, nb, row_ptr, Nn, tot);
    finalize_rowptr_kernel<<<(Nn + 255) / 256, 256, 0, stream>>>(scanned, bsum, row_ptr, cursor, Nn);
    fill_csr_kernel<<<(tot + 255) / 256, 256, 0, stream>>>(src32, dst32, normw, cursor, csr, tot);

    // --- K=10 propagation steps (ping-pong; final step lands in d_out) ---
    const int sgrid = (Nn * 32 + 255) / 256;   // one wave (32 lanes) per node
    float* bufs[2] = { bufA, bufB };
    const float* hin = z;                       // h0 = z
    for (int k = 0; k < 10; ++k) {
        float* hout = (k == 9) ? (float*)d_out : bufs[k & 1];
        appnp_step_kernel<<<sgrid, 256, 0, stream>>>(row_ptr, csr, z, hin, hout, Nn);
        hin = hout;
    }
}
